// SVGMDNTop_56332791054986
// MI455X (gfx1250) — matrix operand order repaired and checked by reading, not compiled, not run on hardware
//
#include <hip/hip_runtime.h>
#include <hip/hip_bf16.h>

typedef __attribute__((ext_vector_type(16))) __bf16 v16bf;
typedef __attribute__((ext_vector_type(8)))  __bf16 v8bf;
typedef __attribute__((ext_vector_type(8)))  float  v8f;

#define H_DIM   1024
#define NOUT    904          // CMD_LEN + 3*NUM_MIX*ARG_LEN
#define NPAD    1024         // N padded to 64 tiles of 16
#define CMDL    4
#define NMIX    50
#define ARGL    6
#define BM      32           // M rows per block (2 subtiles of 16 per wave)
#define KC      128          // K chunk staged in LDS per barrier
#define NT      8            // N-tiles (of 16) per wave

// ---------------------------------------------------------------------------
// Kernel 0: W [904 x 1024] f32 -> bf16, zero-padded to [1024 x 1024] in ws.
// ---------------------------------------------------------------------------
__global__ __launch_bounds__(256) void convert_W_kernel(
    const float* __restrict__ W, __bf16* __restrict__ Wb) {
  int idx = blockIdx.x * blockDim.x + threadIdx.x;   // 0 .. NPAD*H-1
  int n = idx >> 10;
  int k = idx & (H_DIM - 1);
  float v = (n < NOUT) ? W[n * H_DIM + k] : 0.0f;
  Wb[idx] = (__bf16)v;
}

__device__ inline v8bf pack8(float4 a, float4 b) {
  v8bf p;
  p[0] = (__bf16)a.x; p[1] = (__bf16)a.y; p[2] = (__bf16)a.z; p[3] = (__bf16)a.w;
  p[4] = (__bf16)b.x; p[5] = (__bf16)b.y; p[6] = (__bf16)b.z; p[7] = (__bf16)b.w;
  return p;
}

// ---------------------------------------------------------------------------
// Kernel 1: fused GEMM (bf16 WMMA, f32 accum) + bias + argmax sampling.
// 256 threads = 8 waves, one per N-group (128 cols each).
// Each wave computes TWO 16x128 M-subtiles (rows 0-15 and 16-31), so every
// B fragment loaded from L2 feeds two WMMAs (1 b128-load-pair : 2 wmma).
// A K-chunk (32x128) is double-buffered in LDS; one barrier per chunk.
// Reductions: temp = 1e-4 => categorical == argmax; the JAX-RNG Gaussian
// term (scale 0.01*exp(logstd)) is dropped for determinism.
// ---------------------------------------------------------------------------
__global__ __launch_bounds__(256) void mdn_fused_kernel(
    const float*  __restrict__ A,     // [M][1024] fp32
    const __bf16* __restrict__ Wb,    // [1024][1024] bf16 (N-major, padded)
    const float*  __restrict__ bias,  // [904]
    float*        __restrict__ out)   // [M][10]
{
  __shared__ __align__(16) float  sC[BM * NOUT];     // 113 KB result tile
  __shared__ __align__(16) __bf16 sA[2][BM * KC];    // 2 x 8 KB A chunks

  const int t    = threadIdx.x;
  const int lane = t & 31;
  const int ng   = t >> 5;             // wave id == N-group, 0..7
  const int lh   = lane >> 4;          // lane half
  const int l15  = lane & 15;
  const int mbase = blockIdx.x * BM;

  v8f acc0[NT] = {};                   // rows 0..15  of the block tile
  v8f acc1[NT] = {};                   // rows 16..31 of the block tile

  // per-thread A staging coords: 16 consecutive floats of one row
  const int r0 = t >> 3;               // 0..31
  const int c0 = (t & 7) * 16;         // 0..112
  const float* aRow = A + (size_t)(mbase + r0) * H_DIM + c0;

  // ---- stage chunk 0 ----
  {
    const float4 f0 = *(const float4*)(aRow + 0);
    const float4 f1 = *(const float4*)(aRow + 4);
    const float4 f2 = *(const float4*)(aRow + 8);
    const float4 f3 = *(const float4*)(aRow + 12);
    *(v8bf*)(&sA[0][r0 * KC + c0])     = pack8(f0, f1);
    *(v8bf*)(&sA[0][r0 * KC + c0 + 8]) = pack8(f2, f3);
  }
  __syncthreads();

  for (int oc = 0; oc < H_DIM / KC; ++oc) {          // 8 chunks
    const __bf16* cur = sA[oc & 1];
    const bool more = (oc + 1) < (H_DIM / KC);

    // issue next-chunk global loads early (overlap with compute below)
    float4 n0, n1, n2, n3;
    if (more) {
      n0 = *(const float4*)(aRow + (oc + 1) * KC);
      n1 = *(const float4*)(aRow + (oc + 1) * KC + 4);
      n2 = *(const float4*)(aRow + (oc + 1) * KC + 8);
      n3 = *(const float4*)(aRow + (oc + 1) * KC + 12);
      if (oc + 2 < (H_DIM / KC))                      // warm L2 two ahead
        __builtin_prefetch(aRow + (oc + 2) * KC, 0, 3);
    }

#pragma unroll
    for (int ks = 0; ks < KC / 32; ++ks) {           // 4 sub-steps of K=32
      // A fragments for both M-subtiles (16x32 each):
      // lanes 0-15 hold K{0..7,16..23} of row M=l15; lanes 16-31 K{8..15,24..31}.
      union { v16bf v; v8bf h[2]; } af0, af1;
      {
        const __bf16* ap0 = cur + l15 * KC + ks * 32 + lh * 8;
        af0.h[0] = *(const v8bf*)(ap0);
        af0.h[1] = *(const v8bf*)(ap0 + 16);
        const __bf16* ap1 = ap0 + 16 * KC;
        af1.h[0] = *(const v8bf*)(ap1);
        af1.h[1] = *(const v8bf*)(ap1 + 16);
      }
      // Each B fragment (32B/lane from L2) feeds TWO WMMAs (both M-subtiles).
#pragma unroll
      for (int tn = 0; tn < NT; ++tn) {
        const int n = ((ng * NT + tn) << 4) + l15;
        union { v16bf v; v8bf h[2]; } bf;
        const __bf16* bp =
            Wb + (size_t)n * H_DIM + oc * KC + ks * 32 + lh * 8;
        bf.h[0] = *(const v8bf*)(bp);
        bf.h[1] = *(const v8bf*)(bp + 16);
        acc0[tn] = __builtin_amdgcn_wmma_f32_16x16x32_bf16(
            false, af0.v, false, bf.v, (short)0, acc0[tn], false, false);
        acc1[tn] = __builtin_amdgcn_wmma_f32_16x16x32_bf16(
            false, af1.v, false, bf.v, (short)0, acc1[tn], false, false);
      }
    }

    // convert + store next chunk into the other LDS buffer, then one barrier
    if (more) {
      *(v8bf*)(&sA[(oc + 1) & 1][r0 * KC + c0])     = pack8(n0, n1);
      *(v8bf*)(&sA[(oc + 1) & 1][r0 * KC + c0 + 8]) = pack8(n2, n3);
    }
    __syncthreads();
  }

  // ---- epilogue: add bias, spill C tiles to LDS (drop padded n >= 904) ----
  // C/D layout: VGPR v of lane holds (M = v + 8*lane_half, N = lane&15).
#pragma unroll
  for (int tn = 0; tn < NT; ++tn) {
    const int n = ((ng * NT + tn) << 4) + l15;
    if (n < NOUT) {
      const float bv = bias[n];
#pragma unroll
      for (int v = 0; v < 8; ++v) {
        const int m = v + lh * 8;
        sC[m * NOUT + n]        = acc0[tn][v] + bv;
        sC[(16 + m) * NOUT + n] = acc1[tn][v] + bv;
      }
    }
  }
  __syncthreads();

  // ---- command head: one thread per row, argmax over 4 logits -> one-hot ---
  if (t < BM) {
    const float* row = sC + t * NOUT;
    int   best = 0;
    float bv   = row[0];
#pragma unroll
    for (int c = 1; c < CMDL; ++c) {
      const float x = row[c];
      if (x > bv) { bv = x; best = c; }
    }
    float* o = out + (size_t)(mbase + t) * (CMDL + ARGL);
#pragma unroll
    for (int c = 0; c < CMDL; ++c) o[c] = (c == best) ? 1.0f : 0.0f;
  }

  // ---- mixture heads: 32 rows * 6 slots = 192 tasks.
  //      argmax over 50 logmix (logsumexp shift cannot change the argmax),
  //      then gather the corresponding mean. ----
  if (t < BM * ARGL) {
    const int r = t / ARGL, j = t % ARGL;
    const float* grp = sC + r * NOUT + CMDL + j * (3 * NMIX);
    int   best = 0;
    float bv   = grp[0];
    for (int i = 1; i < NMIX; ++i) {
      const float x = grp[i];
      if (x > bv) { bv = x; best = i; }
    }
    out[(size_t)(mbase + r) * (CMDL + ARGL) + CMDL + j] = grp[NMIX + best];
  }
}

// ---------------------------------------------------------------------------
extern "C" void kernel_launch(void* const* d_in, const int* in_sizes, int n_in,
                              void* d_out, int out_size, void* d_ws,
                              size_t ws_size, hipStream_t stream) {
  (void)in_sizes; (void)n_in; (void)out_size; (void)ws_size;

  const float* A    = (const float*)d_in[0];  // decoder_output [51,2048,1024]
  const float* W    = (const float*)d_in[1];  // [904,1024]
  const float* bias = (const float*)d_in[2];  // [904]
  float*       out  = (float*)d_out;          // [51,2048,10]
  __bf16*      Wb   = (__bf16*)d_ws;          // 2 MB scratch: bf16 W, padded

  convert_W_kernel<<<(NPAD * H_DIM) / 256, 256, 0, stream>>>(W, Wb);

  const int M = 51 * 2048;                    // 104448 = 3264 * 32
  mdn_fused_kernel<<<M / BM, 256, 0, stream>>>(A, Wb, bias, out);
}
